// SelfAttention2_28561532518886
// MI455X (gfx1250) — matrix operand order repaired and checked by reading, not compiled, not run on hardware
//
#include <hip/hip_runtime.h>
#include <hip/hip_bf16.h>

// ---------------------------------------------------------------------------
// SelfAttention2: q,k,v = x@W^T + b ; P = sigmoid(k_i . q_j) * m_i * m_j
//                 out_i = sum_j P[i][j] * v_j      (note transposed scores!)
// bf16 WMMA (v_wmma_f32_16x16x32_bf16), f32 accumulation, fused attention,
// async global->LDS staging (ASYNCcnt) with double buffering.
// ---------------------------------------------------------------------------

typedef __attribute__((ext_vector_type(16))) __bf16 v16bf;
typedef __attribute__((ext_vector_type(8)))  float  v8f;
typedef int v4i __attribute__((vector_size(16)));

static constexpr int B_ = 4;
static constexpr int S_ = 2048;
static constexpr int D_ = 512;   // = H*K = H*V
static constexpr int H_ = 8;
static constexpr int HK = 64;    // head dim

#if defined(__has_builtin)
#if __has_builtin(__builtin_amdgcn_global_load_async_to_lds_b128) && \
    __has_builtin(__builtin_amdgcn_s_wait_asynccnt)
#define HAVE_ASYNC 1
#endif
#endif
#ifndef HAVE_ASYNC
#define HAVE_ASYNC 0
#endif

#if HAVE_ASYNC
// builtin expects (v4i addrspace(1)* src, v4i addrspace(3)* dst, imm off, imm cpol)
// low 32 bits of a generic LDS address == LDS byte offset (aperture mapping)
#define GAS(p) ((__attribute__((address_space(1))) v4i*)(unsigned long long)(size_t)(p))
#define LAS(p) ((__attribute__((address_space(3))) v4i*)(unsigned int)(size_t)(p))
#endif

// float -> bf16 bits, round-to-nearest-even
__device__ __forceinline__ unsigned short f32_to_bf16(float f) {
    unsigned int u = __builtin_bit_cast(unsigned int, f);
    u += 0x7fffu + ((u >> 16) & 1u);
    return (unsigned short)(u >> 16);
}
__device__ __forceinline__ unsigned int pack_bf16(float lo, float hi) {
    return (unsigned int)f32_to_bf16(lo) | ((unsigned int)f32_to_bf16(hi) << 16);
}

union Frag { v16bf v; unsigned int u[8]; };

// A-fragment K-pair base per VGPR index v (16-bit A 16x32 layout, ISA 7.12.2)
__device__ __forceinline__ int a_kpair(int v, int hh) {
    return ((v < 4) ? 2 * v : 16 + 2 * (v - 4)) + hh * 8;
}

// ---------------------------------------------------------------------------
// Projection GEMM: Out[m,n] = bf16( (sum_d X[m,d]*W[n,d] + bias[n]) * scale )
// WG tile 128x64, 8 waves as 4(M) x 2(N), each wave 32x32 (4 accums).
// ---------------------------------------------------------------------------
__global__ __launch_bounds__(256)
void qkv_gemm(const float* __restrict__ X, const float* __restrict__ W,
              const float* __restrict__ bias, float scale,
              unsigned short* __restrict__ Out, int M, int N, int Kd)
{
    __shared__ unsigned short As[128 * 40];   // [m][k], stride 40 (pad)
    __shared__ unsigned short Bs[32 * 72];    // [k][n], stride 72 (pad)

    const int tid  = threadIdx.x;
    const int lane = tid & 31;
    const int w    = tid >> 5;     // wave 0..7
    const int wm   = w >> 1;       // 0..3  (M sub-tile)
    const int wn   = w & 1;        // 0..1  (N sub-tile)
    const int r    = lane & 15;
    const int hh   = lane >> 4;

    const int m0 = blockIdx.x * 128;
    const int n0 = blockIdx.y * 64;

    v8f acc[2][2] = {};

    for (int kt = 0; kt < Kd; kt += 32) {
        __syncthreads();
        {   // stage A tile 128x32: float4 loads, packed uint2 LDS stores
            int row = tid >> 1;
            int cb  = (tid & 1) * 16;
            const float* src = X + (size_t)(m0 + row) * Kd + kt + cb;
            unsigned short* dst = As + row * 40 + cb;
            #pragma unroll
            for (int j = 0; j < 4; ++j) {
                float4 f = *(const float4*)(src + 4 * j);
                uint2 p;
                p.x = pack_bf16(f.x, f.y);
                p.y = pack_bf16(f.z, f.w);
                *(uint2*)&dst[4 * j] = p;
            }
        }
        {   // stage W tile 64x32 transposed into Bs[k][n] (strided b16 stores)
            int n  = tid >> 2;           // 0..63
            int kb = (tid & 3) * 8;      // 0,8,16,24
            const float* src = W + (size_t)(n0 + n) * Kd + kt + kb;
            float4 f0 = ((const float4*)src)[0];
            float4 f1 = ((const float4*)src)[1];
            float e[8] = {f0.x, f0.y, f0.z, f0.w, f1.x, f1.y, f1.z, f1.w};
            #pragma unroll
            for (int j = 0; j < 8; ++j) Bs[(kb + j) * 72 + n] = f32_to_bf16(e[j]);
        }
        __syncthreads();

        Frag a[2], bf[2];
        #pragma unroll
        for (int f = 0; f < 2; ++f) {
            int row = wm * 32 + f * 16 + r;
            #pragma unroll
            for (int v = 0; v < 8; ++v)
                a[f].u[v] = *(const unsigned int*)&As[row * 40 + a_kpair(v, hh)];
        }
        #pragma unroll
        for (int g = 0; g < 2; ++g) {
            int kin = hh * 16 + r;
            int cb  = wn * 32 + g * 16;
            #pragma unroll
            for (int v = 0; v < 8; ++v)
                bf[g].u[v] = (unsigned int)Bs[a_kpair(v, hh) * 72 + cb + r] | ((unsigned int)Bs[(a_kpair(v, hh) + 1) * 72 + cb + r] << 16);
        }
        #pragma unroll
        for (int f = 0; f < 2; ++f)
            #pragma unroll
            for (int g = 0; g < 2; ++g)
                acc[f][g] = __builtin_amdgcn_wmma_f32_16x16x32_bf16(
                    false, a[f].v, false, bf[g].v, (short)0, acc[f][g], false, false);
    }

    float bb[2];
    #pragma unroll
    for (int g = 0; g < 2; ++g) bb[g] = bias[n0 + wn * 32 + g * 16 + r];

    #pragma unroll
    for (int f = 0; f < 2; ++f)
        #pragma unroll
        for (int g = 0; g < 2; ++g)
            #pragma unroll
            for (int v = 0; v < 8; ++v) {
                int m = m0 + wm * 32 + f * 16 + hh * 8 + v;   // C/D layout
                int n = n0 + wn * 32 + g * 16 + r;
                Out[(size_t)m * N + n] = f32_to_bf16((acc[f][g][v] + bb[g]) * scale);
            }
}

// ---------------------------------------------------------------------------
// Fused sigmoid-attention (transposed): out_i = sum_j sig(k_i.q_j)*m_i*m_j*v_j
// Left = Kb rows (i), Right = Qb cols (j), values = Vb rows (j).
// WG: (b,h, 128-row i-tile); wave owns 16 rows; j-loop in tiles of 32,
// double-buffered LDS, V tile staged with GLOBAL_LOAD_ASYNC_TO_LDS_B128.
// ---------------------------------------------------------------------------
__global__ __launch_bounds__(256)
void attn(const unsigned short* __restrict__ Kb,   // bf16 [B*S, D] left rows
          const unsigned short* __restrict__ Qb,   // bf16 [B*S, D] right cols
          const unsigned short* __restrict__ Vb,   // bf16 [B*S, D] values
          const float* __restrict__ mask,          // [B, S]
          float* __restrict__ out)                 // [B*S, D]
{
    __shared__ unsigned short QT[2][64 * 34];     // [c(0..63)][j(0..31)] transposed
    __shared__ unsigned short Vs[2][32 * 72];     // [j][vv] row-major
    __shared__ unsigned short Ps[8][16 * 32];     // per-wave P tile re-layout

    const int tid  = threadIdx.x;
    const int lane = tid & 31;
    const int w    = tid >> 5;
    const int r    = lane & 15;
    const int hh   = lane >> 4;

    const int bh = blockIdx.y;
    const int b  = bh >> 3;
    const int h  = bh & 7;
    const int i0 = blockIdx.x * 128 + w * 16;
    const size_t rowBase = (size_t)b * S_;

    // staging geometry (whole workgroup): 32 rows(j) x 64 cols, 16B per thread
    const int jr = tid >> 3;          // 0..31
    const int cb = (tid & 7) * 8;     // 0..56

    auto stage = [&](int jt, int buf) {
        const int j0 = jt * 32;
        // Q^T tile: b128 global load, transposed b16 LDS stores
        const unsigned short* qsrc = Qb + (rowBase + j0 + jr) * D_ + h * HK + cb;
        union { uint4 q; unsigned short s[8]; } t;
        t.q = *(const uint4*)qsrc;
        #pragma unroll
        for (int e = 0; e < 8; ++e) QT[buf][(cb + e) * 34 + jr] = t.s[e];
        // V tile: contiguous 16B copy -> async global->LDS if available
        const unsigned short* vsrc = Vb + (rowBase + j0 + jr) * D_ + h * HK + cb;
        unsigned short* vdst = &Vs[buf][jr * 72 + cb];
#if HAVE_ASYNC
        __builtin_amdgcn_global_load_async_to_lds_b128(GAS(vsrc), LAS(vdst), 0, 0);
#else
        *(uint4*)vdst = *(const uint4*)vsrc;
#endif
    };

    // left (K-vector) A-fragments for this wave's 16 rows, head dim 64 = 2 steps
    Frag lf[2];
    #pragma unroll
    for (int st = 0; st < 2; ++st)
        #pragma unroll
        for (int v = 0; v < 8; ++v) {
            int k0 = a_kpair(v, hh) + st * 32;
            lf[st].u[v] = *(const unsigned int*)(Kb + (rowBase + i0 + r) * D_ + h * HK + k0);
        }

    float rmask[8];
    #pragma unroll
    for (int v = 0; v < 8; ++v) rmask[v] = mask[rowBase + i0 + hh * 8 + v];

    v8f o[4] = {};

    constexpr int NT = S_ / 32;       // 64 j-tiles
    stage(0, 0);
#if HAVE_ASYNC
    __builtin_amdgcn_s_wait_asynccnt(0);
#endif
    __syncthreads();

    for (int jt = 0; jt < NT; ++jt) {
        const int buf = jt & 1;
        if (jt + 1 < NT) stage(jt + 1, buf ^ 1);   // overlap next tile with compute

        const int j0 = jt * 32;
        const float cmask0 = mask[rowBase + j0 + r];
        const float cmask1 = mask[rowBase + j0 + 16 + r];

        #pragma unroll
        for (int jc = 0; jc < 2; ++jc) {
            Frag bq[2];
            #pragma unroll
            for (int st = 0; st < 2; ++st) {
                int kin = st * 32 + hh * 16 + r;
                #pragma unroll
                for (int v = 0; v < 8; ++v)
                    bq[st].u[v] = (unsigned int)QT[buf][(st * 32 + a_kpair(v, hh)) * 34 + jc * 16 + r] | ((unsigned int)QT[buf][(st * 32 + a_kpair(v, hh) + 1) * 34 + jc * 16 + r] << 16);
            }
            v8f s = {};
            s = __builtin_amdgcn_wmma_f32_16x16x32_bf16(false, lf[0].v, false, bq[0].v, (short)0, s, false, false);
            s = __builtin_amdgcn_wmma_f32_16x16x32_bf16(false, lf[1].v, false, bq[1].v, (short)0, s, false, false);

            const float cm = (jc == 0) ? cmask0 : cmask1;
            #pragma unroll
            for (int v = 0; v < 8; ++v) {
                float p = 1.f / (1.f + __expf(-s[v]));        // sigmoid (no softmax)
                p *= rmask[v] * cm;
                Ps[w][(hh * 8 + v) * 32 + jc * 16 + r] = f32_to_bf16(p);
            }
        }

        // re-layout P (accumulator layout) -> A-fragment via per-wave LDS patch
        Frag pf;
        #pragma unroll
        for (int v = 0; v < 8; ++v)
            pf.u[v] = *(const unsigned int*)&Ps[w][r * 32 + a_kpair(v, hh)];

        #pragma unroll
        for (int g = 0; g < 4; ++g) {
            Frag bv;
            int kin = hh * 16 + r;
            #pragma unroll
            for (int v = 0; v < 8; ++v)
                bv.u[v] = (unsigned int)Vs[buf][a_kpair(v, hh) * 72 + g * 16 + r] | ((unsigned int)Vs[buf][(a_kpair(v, hh) + 1) * 72 + g * 16 + r] << 16);
            o[g] = __builtin_amdgcn_wmma_f32_16x16x32_bf16(
                false, pf.v, false, bv.v, (short)0, o[g], false, false);
        }

#if HAVE_ASYNC
        __builtin_amdgcn_s_wait_asynccnt(0);      // next tile's V is in LDS
#endif
        __syncthreads();
    }

    #pragma unroll
    for (int g = 0; g < 4; ++g)
        #pragma unroll
        for (int v = 0; v < 8; ++v) {
            int m = i0 + hh * 8 + v;
            int c = h * HK + g * 16 + r;
            out[(rowBase + m) * D_ + c] = o[g][v];
        }
}

// ---------------------------------------------------------------------------
extern "C" void kernel_launch(void* const* d_in, const int* in_sizes, int n_in,
                              void* d_out, int out_size, void* d_ws, size_t ws_size,
                              hipStream_t stream) {
    const float* x    = (const float*)d_in[0];
    const float* mask = (const float*)d_in[1];
    const float* Wq   = (const float*)d_in[2];
    const float* bq   = (const float*)d_in[3];
    const float* Wk   = (const float*)d_in[4];
    const float* bk   = (const float*)d_in[5];
    const float* Wv   = (const float*)d_in[6];
    const float* bv   = (const float*)d_in[7];
    float* out = (float*)d_out;

    const size_t elems = (size_t)B_ * S_ * D_;       // 4,194,304
    unsigned short* Qb = (unsigned short*)d_ws;      // bf16 bits, 8 MB each
    unsigned short* Kb = Qb + elems;
    unsigned short* Vb = Kb + elems;

    const int M = B_ * S_;                           // 8192
    dim3 gg(M / 128, D_ / 64);                       // 64 x 8 WGs
    qkv_gemm<<<gg, 256, 0, stream>>>(x, Wq, bq, 1.0f,   Qb, M, D_, D_);
    qkv_gemm<<<gg, 256, 0, stream>>>(x, Wk, bk, 0.125f, Kb, M, D_, D_);  // sqrt(1/64)
    qkv_gemm<<<gg, 256, 0, stream>>>(x, Wv, bv, 1.0f,   Vb, M, D_, D_);

    dim3 ga(S_ / 128, B_ * H_);                      // 16 x 32 WGs
    attn<<<ga, 256, 0, stream>>>(Kb, Qb, Vb, mask, out);
}
